// PermuterGatingUnit_53008486367564
// MI455X (gfx1250) — matrix operand order repaired and checked by reading, not compiled, not run on hardware
//
#include <hip/hip_runtime.h>
#include <hip/hip_bf16.h>

// ---------------------------------------------------------------------------
// Types for CDNA5 WMMA (wave32): bf16 A/B fragments are 16 x bf16 (8 VGPRs),
// f32 C/D accumulators are 8 x f32 (8 VGPRs).
// ---------------------------------------------------------------------------
typedef __attribute__((ext_vector_type(16))) __bf16 v16bf;
typedef __attribute__((ext_vector_type(8)))  float  v8f;
typedef __attribute__((ext_vector_type(8)))  unsigned int v8u;

#define B_DIM      8
#define T_DIM      4096
#define D_MODEL    512
#define HEADS      8
#define D_HEAD     64
#define BUCKET     64
#define NB         (T_DIM / BUCKET)     // 64 buckets
#define D_FF       (4 * D_MODEL)        // 2048
#define M_ROWS     (B_DIM * T_DIM)      // 32768

__device__ __forceinline__ unsigned short f32_to_bf16(float f) {
    unsigned int u = __float_as_uint(f);
    unsigned int r = (u + 0x7FFFu + ((u >> 16) & 1u)) >> 16;   // RNE
    return (unsigned short)r;
}

// ---------------------------------------------------------------------------
// Elementwise f32 -> bf16 conversion (activations)
// ---------------------------------------------------------------------------
__global__ void cvt_f32_bf16_kernel(const float* __restrict__ in,
                                    unsigned short* __restrict__ out, long n) {
    long i = (long)blockIdx.x * blockDim.x + threadIdx.x;
    long stride = (long)gridDim.x * blockDim.x;
    for (; i < n; i += stride) out[i] = f32_to_bf16(in[i]);
}

// ---------------------------------------------------------------------------
// One-time weight swizzle: W[K][N] f32 -> bf16 pairs laid out in WMMA B
// fragment order, tiled as contiguous 2048-dword blocks per (n-block, k-tile):
//   out[((bnb*KT + kt)*2048) + (tn*32 + lane)*8 + g]
//   lane = half*16 + (n%16), half = (k%16)/8, g = ((k%32)/16)*4 + (k%8)/2
// so the GEMM's B staging becomes a contiguous async copy.
// ---------------------------------------------------------------------------
__global__ void swizzle_w_kernel(const float* __restrict__ W,
                                 unsigned int* __restrict__ out, int K, int N) {
    const int KT = K >> 5;
    const long total = (long)KT * (N >> 7) * 2048;
    for (long s = (long)blockIdx.x * blockDim.x + threadIdx.x; s < total;
         s += (long)gridDim.x * blockDim.x) {
        int tile   = (int)(s >> 11);
        int within = (int)(s & 2047);
        int tn   = within >> 8;            // 0..7
        int lane = (within >> 3) & 31;     // 0..31
        int g    = within & 7;             // 0..7
        int kt   = tile % KT;
        int bnb  = tile / KT;
        int half = lane >> 4, r = lane & 15;
        int k = kt * 32 + (g >> 2) * 16 + half * 8 + (g & 3) * 2;
        int n = bnb * 128 + tn * 16 + r;
        unsigned lo = f32_to_bf16(W[(long)k * N + n]);
        unsigned hi = f32_to_bf16(W[(long)(k + 1) * N + n]);
        out[s] = lo | (hi << 16);
    }
}

// ---------------------------------------------------------------------------
// LayerNorm over last dim (D=512), one row per block, bf16 output.
// ---------------------------------------------------------------------------
__global__ __launch_bounds__(256)
void layernorm_bf16_kernel(const float* __restrict__ x,
                           const float* __restrict__ g,
                           const float* __restrict__ b,
                           unsigned short* __restrict__ out) {
    const int D = D_MODEL;
    const float* row = x + (long)blockIdx.x * D;
    __shared__ float red[256];
    const int tid = threadIdx.x;

    float s = 0.f;
    for (int i = tid; i < D; i += 256) s += row[i];
    red[tid] = s; __syncthreads();
    for (int off = 128; off > 0; off >>= 1) {
        if (tid < off) red[tid] += red[tid + off];
        __syncthreads();
    }
    float mu = red[0] * (1.0f / D);
    __syncthreads();

    float v = 0.f;
    for (int i = tid; i < D; i += 256) { float d = row[i] - mu; v += d * d; }
    red[tid] = v; __syncthreads();
    for (int off = 128; off > 0; off >>= 1) {
        if (tid < off) red[tid] += red[tid + off];
        __syncthreads();
    }
    float rstd = rsqrtf(red[0] * (1.0f / D) + 1e-5f);

    unsigned short* orow = out + (long)blockIdx.x * D;
    for (int i = tid; i < D; i += 256)
        orow[i] = f32_to_bf16((row[i] - mu) * rstd * g[i] + b[i]);
}

// ---------------------------------------------------------------------------
// bf16 WMMA GEMM: C = act(A[MxK] * Bsw + bias) (+resid) (*mul)
// Block tile 128x128, 8 waves (4x2), each wave 32x64 = 2x4 WMMA 16x16 tiles,
// K-step 32 via V_WMMA_F32_16X16X32_BF16.
//
// Data movement: both A (bf16 row-major; every aligned 8-K chunk of a row maps
// to 4 contiguous LDS dwords of the fragment layout) and B (pre-swizzled
// contiguous tiles) are staged with GLOBAL_LOAD_ASYNC_TO_LDS_B128 (ASYNCcnt),
// double-buffered: next tile's 4 async ops stay in flight during WMMA issue.
// ---------------------------------------------------------------------------
template<int GELU>
__global__ __launch_bounds__(256)
void wmma_gemm_kernel(const unsigned short* __restrict__ A,   // MxK bf16
                      const unsigned int*  __restrict__ Bsw,  // swizzled tiles
                      const float* __restrict__ bias,         // N or null
                      const float* __restrict__ resid,        // MxN or null
                      const float* __restrict__ mulin,        // MxN or null
                      float* __restrict__ outF,               // MxN or null
                      unsigned short* __restrict__ outH,      // MxN or null
                      int M, int N, int K) {
    __shared__ __attribute__((aligned(16))) unsigned int lA[2][2048];
    __shared__ __attribute__((aligned(16))) unsigned int lB[2][2048];

    const int tid  = threadIdx.x;
    const int lane = tid & 31;
    const int wave = tid >> 5;
    const int wm   = wave >> 1;      // 0..3 : 32-row slab
    const int wn   = wave & 1;       // 0..1 : 64-col slab
    const int bm   = blockIdx.y * 128;
    const int bn   = blockIdx.x * 128;
    const int KT   = K >> 5;

    const unsigned ldsA0 = (unsigned)(uintptr_t)&lA[0][0];
    const unsigned ldsB0 = (unsigned)(uintptr_t)&lB[0][0];

    // A: thread copies two 16B chunks; chunk u (0..511): row m=u>>2, q=u&3.
    // LDS dst dwords: ((m>>4)*32 + (q&1)*16 + (m&15))*8 + (q>>1)*4 .. +3
    auto issueA = [&](int kt, int buf) {
#pragma unroll
        for (int t = 0; t < 2; ++t) {
            int u = t * 256 + tid;
            int m = u >> 2, q = u & 3;
            const void* g = (const void*)(A + (long)(bm + m) * K + kt * 32 + q * 8);
            unsigned lds = ldsA0 + (unsigned)buf * 8192u +
                ((((m >> 4) * 32) + (q & 1) * 16 + (m & 15)) * 8 + (q >> 1) * 4) * 4u;
            asm volatile("global_load_async_to_lds_b128 %0, %1, off"
                         :: "v"(lds), "v"(g) : "memory");
        }
    };
    // B: contiguous copy of the pre-swizzled 2048-dword tile.
    auto issueB = [&](int kt, int buf) {
        const unsigned int* src = Bsw + ((long)blockIdx.x * KT + kt) * 2048;
#pragma unroll
        for (int t = 0; t < 2; ++t) {
            int c = t * 256 + tid;
            const void* g = (const void*)(src + c * 4);
            unsigned lds = ldsB0 + (unsigned)buf * 8192u + (unsigned)c * 16u;
            asm volatile("global_load_async_to_lds_b128 %0, %1, off"
                         :: "v"(lds), "v"(g) : "memory");
        }
    };

    v8f acc[2][4];
    const v8f zero = {0.f, 0.f, 0.f, 0.f, 0.f, 0.f, 0.f, 0.f};
#pragma unroll
    for (int i = 0; i < 2; ++i)
#pragma unroll
        for (int j = 0; j < 4; ++j) acc[i][j] = zero;

    issueA(0, 0);
    issueB(0, 0);
    for (int kt = 0; kt < KT; ++kt) {
        const int cur = kt & 1;
        if (kt + 1 < KT) {
            issueA(kt + 1, cur ^ 1);           // overlap next tile's DMA
            issueB(kt + 1, cur ^ 1);
            asm volatile("s_wait_asynccnt 0x4" ::: "memory");  // cur tile done
        } else {
            asm volatile("s_wait_asynccnt 0x0" ::: "memory");
        }
        __syncthreads();

        v16bf af[2], bf[4];
#pragma unroll
        for (int i = 0; i < 2; ++i) {
            int tm = wm * 2 + i;
            v8u raw = *(const v8u*)&lA[cur][(tm * 32 + lane) * 8];
            af[i] = __builtin_bit_cast(v16bf, raw);
        }
#pragma unroll
        for (int j = 0; j < 4; ++j) {
            int tn = wn * 4 + j;
            v8u raw = *(const v8u*)&lB[cur][(tn * 32 + lane) * 8];
            bf[j] = __builtin_bit_cast(v16bf, raw);
        }
#pragma unroll
        for (int i = 0; i < 2; ++i)
#pragma unroll
            for (int j = 0; j < 4; ++j)
                acc[i][j] = __builtin_amdgcn_wmma_f32_16x16x32_bf16(
                    false, af[i], false, bf[j], (short)0, acc[i][j],
                    false, false);
        __syncthreads();   // safe to overwrite this buffer two tiles later
    }

    // ---- epilogue: D layout = VGPR v -> M = v (+8 for lanes 16..31) ----
    const int nl = lane & 15;
    const int mh = lane >> 4;
#pragma unroll
    for (int i = 0; i < 2; ++i) {
#pragma unroll
        for (int j = 0; j < 4; ++j) {
            int mbase = bm + wm * 32 + i * 16 + mh * 8;
            int n     = bn + wn * 64 + j * 16 + nl;
            union { v8f v; float f[8]; } cu; cu.v = acc[i][j];
            float bi = bias ? bias[n] : 0.f;
#pragma unroll
            for (int v = 0; v < 8; ++v) {
                long idx = (long)(mbase + v) * N + n;
                float val = cu.f[v] + bi;
                if (GELU) val = 0.5f * val * (1.f + erff(val * 0.70710678f));
                if (resid) val += resid[idx];
                if (mulin) val *= mulin[idx];
                if (outF) outF[idx] = val;
                if (outH) outH[idx] = f32_to_bf16(val);
            }
        }
    }
}

// ---------------------------------------------------------------------------
// AttentionSortNet: bucket means -> R = sq sk^T * scale -> differentiable
// top-2 (softmax prob at iterated argmax). One block per bh (64 blocks).
// ---------------------------------------------------------------------------
__global__ __launch_bounds__(256)
void sortnet_topk_kernel(const float* __restrict__ q,    // (B*T, 512)
                         const float* __restrict__ kv,   // (B*T, 1024)
                         int*   __restrict__ topIdx,     // (64,64,2)
                         float* __restrict__ topVal) {   // (64,64,2)
    const int bh = blockIdx.x;
    const int bI = bh >> 3, h = bh & 7;
    __shared__ float sq[NB * D_HEAD], sk[NB * D_HEAD], R[NB * NB];
    const int tid = threadIdx.x;

    for (int u = tid; u < NB * D_HEAD; u += 256) {
        int j = u >> 6, d = u & 63;
        float aq = 0.f, ak = 0.f;
        long rq = ((long)bI * T_DIM + j * BUCKET) * D_MODEL + h * D_HEAD + d;
        long rk = ((long)bI * T_DIM + j * BUCKET) * (2 * D_MODEL) + h * D_HEAD + d;
        for (int i = 0; i < BUCKET; ++i) {
            aq += q[rq + (long)i * D_MODEL];
            ak += kv[rk + (long)i * (2 * D_MODEL)];
        }
        sq[u] = aq * (1.f / BUCKET);
        sk[u] = ak * (1.f / BUCKET);
    }
    __syncthreads();
    for (int u = tid; u < NB * NB; u += 256) {
        int i = u >> 6, j = u & 63;
        float s = 0.f;
#pragma unroll 8
        for (int d = 0; d < D_HEAD; ++d) s += sq[i * 64 + d] * sk[j * 64 + d];
        R[u] = s * 0.125f;                  // d_head^-0.5
    }
    __syncthreads();
    if (tid < NB) {
        const float* r = &R[tid * NB];
        float m1 = -1e30f; int i1 = 0;
        for (int j = 0; j < NB; ++j) if (r[j] > m1) { m1 = r[j]; i1 = j; }
        float s1 = 0.f;
        for (int j = 0; j < NB; ++j) s1 += __expf(r[j] - m1);
        float m2 = -1e30f; int i2 = 0;
        for (int j = 0; j < NB; ++j) if (j != i1 && r[j] > m2) { m2 = r[j]; i2 = j; }
        float s2 = 0.f;
        for (int j = 0; j < NB; ++j) if (j != i1) s2 += __expf(r[j] - m2);
        int base = (bh * NB + tid) * 2;
        topIdx[base + 0] = i1;  topVal[base + 0] = 1.f / s1;
        topIdx[base + 1] = i2;  topVal[base + 1] = 1.f / s2;
    }
}

// ---------------------------------------------------------------------------
// Bucketed attention: per (bh, query bucket): gather [w0*K_b0 | w1*K_b1 | K_self]
// (192x64), softmax(q k^T * scale), out = attn v. Output merged-heads bf16.
// Dynamic LDS: 160 KB (ok under 320 KB/WGP).
// ---------------------------------------------------------------------------
__global__ __launch_bounds__(256)
void bucket_attention_kernel(const float* __restrict__ q,
                             const float* __restrict__ kv,
                             const int*   __restrict__ topIdx,
                             const float* __restrict__ topVal,
                             unsigned short* __restrict__ attn_bf) {
    extern __shared__ float smem[];
    float* qs = smem;                 // 64*64
    float* ks = qs + 64 * 64;         // 192*64
    float* vs = ks + 192 * 64;        // 192*64
    float* dt = vs + 192 * 64;        // 64*192

    const int qb = blockIdx.x, bh = blockIdx.y;
    const int bI = bh >> 3, h = bh & 7;
    const int tid = threadIdx.x;

    for (int u = tid; u < 64 * 64; u += 256) {
        int i = u >> 6, d = u & 63;
        qs[u] = q[((long)bI * T_DIM + qb * BUCKET + i) * D_MODEL + h * D_HEAD + d];
    }
    const int base = (bh * NB + qb) * 2;
    const int i0 = topIdx[base], i1 = topIdx[base + 1];
    const float w0 = topVal[base], w1 = topVal[base + 1];

    for (int u = tid; u < 192 * 64; u += 256) {
        int row = u >> 6, d = u & 63;
        int slot = row >> 6, i = row & 63;
        int srcb = (slot == 0) ? i0 : ((slot == 1) ? i1 : qb);
        float w  = (slot == 0) ? w0 : ((slot == 1) ? w1 : 1.f);
        long t = (long)bI * T_DIM + srcb * BUCKET + i;
        ks[u] = w * kv[t * (2 * D_MODEL) + h * D_HEAD + d];
        vs[u] = w * kv[t * (2 * D_MODEL) + D_MODEL + h * D_HEAD + d];
    }
    __syncthreads();

    for (int u = tid; u < 64 * 192; u += 256) {
        int i = u / 192, j = u % 192;
        float s = 0.f;
#pragma unroll 8
        for (int d = 0; d < 64; ++d) s += qs[i * 64 + d] * ks[j * 64 + d];
        dt[u] = s * 0.125f;
    }
    __syncthreads();
    if (tid < 64) {
        float* row = &dt[tid * 192];
        float m = -1e30f;
        for (int j = 0; j < 192; ++j) m = fmaxf(m, row[j]);
        float s = 0.f;
        for (int j = 0; j < 192; ++j) { float e = __expf(row[j] - m); row[j] = e; s += e; }
        float inv = 1.f / s;
        for (int j = 0; j < 192; ++j) row[j] *= inv;
    }
    __syncthreads();
    for (int u = tid; u < 64 * 64; u += 256) {
        int i = u >> 6, d = u & 63;
        float s = 0.f;
#pragma unroll 8
        for (int j = 0; j < 192; ++j) s += dt[i * 192 + j] * vs[j * 64 + d];
        long row = (long)bI * T_DIM + qb * BUCKET + i;
        attn_bf[row * D_MODEL + h * D_HEAD + d] = f32_to_bf16(s);
    }
}

// ---------------------------------------------------------------------------
// Orchestration
// ---------------------------------------------------------------------------
extern "C" void kernel_launch(void* const* d_in, const int* in_sizes, int n_in,
                              void* d_out, int out_size, void* d_ws, size_t ws_size,
                              hipStream_t stream) {
    (void)in_sizes; (void)n_in; (void)out_size; (void)ws_size;
    const float* x     = (const float*)d_in[0];
    const float* Wp    = (const float*)d_in[1];
    const float* bp    = (const float*)d_in[2];
    const float* ln1_g = (const float*)d_in[3];
    const float* ln1_b = (const float*)d_in[4];
    const float* Wq    = (const float*)d_in[5];
    const float* Wkv   = (const float*)d_in[6];
    const float* Wo    = (const float*)d_in[7];
    const float* bo    = (const float*)d_in[8];
    const float* ln2_g = (const float*)d_in[9];
    const float* ln2_b = (const float*)d_in[10];
    const float* W1    = (const float*)d_in[11];
    const float* b1    = (const float*)d_in[12];
    const float* W2    = (const float*)d_in[13];
    const float* b2    = (const float*)d_in[14];

    // ---- workspace carve-up (256B aligned) ----
    char* w = (char*)d_ws;
    auto take = [&](size_t bytes) {
        char* p = w;
        w += (bytes + 255) & ~size_t(255);
        return p;
    };
    const long MD = (long)M_ROWS * D_MODEL;      // 16.8M
    unsigned short* x_bf   = (unsigned short*)take(MD * 2);
    unsigned short* h_bf   = (unsigned short*)take(MD * 2);
    unsigned short* h2_bf  = (unsigned short*)take(MD * 2);
    unsigned short* attnbf = (unsigned short*)take(MD * 2);
    // swizzled weights: K*N bf16 packed as K*N/2 dwords
    unsigned int* wp_sw  = (unsigned int*)take((long)D_MODEL * D_MODEL * 2);
    unsigned int* wq_sw  = (unsigned int*)take((long)D_MODEL * D_MODEL * 2);
    unsigned int* wkv_sw = (unsigned int*)take((long)D_MODEL * 2 * D_MODEL * 2);
    unsigned int* wo_sw  = (unsigned int*)take((long)D_MODEL * D_MODEL * 2);
    unsigned int* w1_sw  = (unsigned int*)take((long)D_MODEL * D_FF * 2);
    unsigned int* w2_sw  = (unsigned int*)take((long)D_FF * D_MODEL * 2);
    float* ubuf  = (float*)take(MD * 4);
    float* qbuf  = (float*)take(MD * 4);
    float* kvbuf = (float*)take(MD * 2 * 4);
    float* ybuf  = (float*)take(MD * 4);
    unsigned short* f1_bf = (unsigned short*)take((long)M_ROWS * D_FF * 2);
    int*   topi = (int*)take((long)64 * NB * 2 * 4);
    float* topv = (float*)take((long)64 * NB * 2 * 4);

    // ---- 1: bf16 conversion of x + fragment-swizzled weights ----
    cvt_f32_bf16_kernel<<<4096, 256, 0, stream>>>(x, x_bf, MD);
    swizzle_w_kernel<<<512,  256, 0, stream>>>(Wp,  wp_sw,  D_MODEL, D_MODEL);
    swizzle_w_kernel<<<512,  256, 0, stream>>>(Wq,  wq_sw,  D_MODEL, D_MODEL);
    swizzle_w_kernel<<<512,  256, 0, stream>>>(Wkv, wkv_sw, D_MODEL, 2 * D_MODEL);
    swizzle_w_kernel<<<512,  256, 0, stream>>>(Wo,  wo_sw,  D_MODEL, D_MODEL);
    swizzle_w_kernel<<<1024, 256, 0, stream>>>(W1,  w1_sw,  D_MODEL, D_FF);
    swizzle_w_kernel<<<1024, 256, 0, stream>>>(W2,  w2_sw,  D_FF,    D_MODEL);

    // ---- 2: ln1 ----
    layernorm_bf16_kernel<<<M_ROWS, 256, 0, stream>>>(x, ln1_g, ln1_b, h_bf);

    // ---- 3: u = x Wp + bp ----
    wmma_gemm_kernel<0><<<dim3(D_MODEL / 128, M_ROWS / 128), 256, 0, stream>>>(
        x_bf, wp_sw, bp, nullptr, nullptr, ubuf, nullptr, M_ROWS, D_MODEL, D_MODEL);

    // ---- 4: q = h Wq ; kv = h Wkv ----
    wmma_gemm_kernel<0><<<dim3(D_MODEL / 128, M_ROWS / 128), 256, 0, stream>>>(
        h_bf, wq_sw, nullptr, nullptr, nullptr, qbuf, nullptr, M_ROWS, D_MODEL, D_MODEL);
    wmma_gemm_kernel<0><<<dim3(2 * D_MODEL / 128, M_ROWS / 128), 256, 0, stream>>>(
        h_bf, wkv_sw, nullptr, nullptr, nullptr, kvbuf, nullptr, M_ROWS, 2 * D_MODEL, D_MODEL);

    // ---- 5: sort-net top-2 + bucketed attention ----
    sortnet_topk_kernel<<<B_DIM * HEADS, 256, 0, stream>>>(qbuf, kvbuf, topi, topv);
    size_t attn_smem = (size_t)(64 * 64 + 2 * 192 * 64 + 64 * 192) * sizeof(float);
    bucket_attention_kernel<<<dim3(NB, B_DIM * HEADS), 256, attn_smem, stream>>>(
        qbuf, kvbuf, topi, topv, attnbf);

    // ---- 6: y = x + attn Wo + bo ----
    wmma_gemm_kernel<0><<<dim3(D_MODEL / 128, M_ROWS / 128), 256, 0, stream>>>(
        attnbf, wo_sw, bo, x, nullptr, ybuf, nullptr, M_ROWS, D_MODEL, D_MODEL);

    // ---- 7: ln2, ff1 (gelu), ff2 fused with residual + final u-multiply ----
    layernorm_bf16_kernel<<<M_ROWS, 256, 0, stream>>>(ybuf, ln2_g, ln2_b, h2_bf);
    wmma_gemm_kernel<1><<<dim3(D_FF / 128, M_ROWS / 128), 256, 0, stream>>>(
        h2_bf, w1_sw, b1, nullptr, nullptr, nullptr, f1_bf, M_ROWS, D_FF, D_MODEL);
    wmma_gemm_kernel<0><<<dim3(D_MODEL / 128, M_ROWS / 128), 256, 0, stream>>>(
        f1_bf, w2_sw, b2, ybuf, ubuf, (float*)d_out, nullptr, M_ROWS, D_MODEL, D_FF);
}
